// GGNN_27092653703672
// MI455X (gfx1250) — matrix operand order repaired and checked by reading, not compiled, not run on hardware
//
#include <hip/hip_runtime.h>
#include <hip/hip_bf16.h>

#define NN   2048
#define SD   32
#define MD   16
#define GD   96   // 3*SD
#define XKP  64   // GRU GEMM K (48) padded to 2*32
#define STEPS 10
#define CH   128  // i-chunk for message kernel LDS tiling

typedef __attribute__((ext_vector_type(16))) _Float16 v16h;
typedef __attribute__((ext_vector_type(8)))  float    v8f;
typedef unsigned int u32x4 __attribute__((ext_vector_type(4)));
typedef int          i32x4 __attribute__((ext_vector_type(4)));
typedef int          i32x8 __attribute__((ext_vector_type(8)));

union V16H { v16h v; _Float16 e[16]; };
union V8F  { v8f  v; float    e[8];  };

#if defined(__has_builtin)
#if __has_builtin(__builtin_amdgcn_tensor_load_to_lds) && __has_builtin(__builtin_amdgcn_s_wait_tensorcnt)
#define HAVE_TDM 1
#endif
#endif

__device__ __forceinline__ float sigmoid_f(float x) { return 1.0f / (1.0f + __expf(-x)); }

#ifdef HAVE_TDM
// Build a D# (ISA cdna5 §8) for a 2-D tile of f32 and issue TENSOR_LOAD_TO_LDS.
//   tile_x elements per row, tile_y rows, row stride stride_x (elements),
//   tensor dims for OOB (we never go OOB).
__device__ __forceinline__ void tdm_load_2d(unsigned int lds_addr, const void* gptr,
                                            unsigned int tile_x, unsigned int tile_y,
                                            unsigned long long stride_x,
                                            unsigned int tdim0, unsigned int tdim1) {
    unsigned long long ga = (unsigned long long)gptr;
    u32x4 g0;
    g0.x = 1u;                                   // count=1, user descriptor
    g0.y = lds_addr;                             // LDS byte address
    g0.z = (unsigned int)ga;                     // global_addr[31:0]
    g0.w = (unsigned int)((ga >> 32) & 0x01FFFFFFu) | 0x80000000u; // addr[56:32] | type=2
    i32x8 g1;
    g1[0] = (int)(2u << 16);                     // workgroup_mask=0, data_size=4B
    g1[1] = (int)((tdim0 & 0xFFFFu) << 16);      // tensor_dim0[15:0] @ bits 63:48
    g1[2] = (int)((tdim0 >> 16) | ((tdim1 & 0xFFFFu) << 16)); // dim0[31:16] | dim1[15:0]
    g1[3] = (int)((tdim1 >> 16) | ((tile_x & 0xFFFFu) << 16)); // dim1[31:16] | tile_dim0
    g1[4] = (int)(tile_y & 0xFFFFu);             // tile_dim1 (tile_dim2 = 0)
    g1[5] = (int)(unsigned int)(stride_x & 0xFFFFFFFFu);       // dim0_stride[31:0]
    g1[6] = (int)(unsigned int)((stride_x >> 32) & 0xFFFFu);   // dim0_stride[47:32]
    g1[7] = 0;
    i32x4 z4 = {0, 0, 0, 0};
#if __clang_major__ >= 23
    i32x8 z8 = {0, 0, 0, 0, 0, 0, 0, 0};
    __builtin_amdgcn_tensor_load_to_lds(g0, g1, z4, z4, z8, 0);
#else
    __builtin_amdgcn_tensor_load_to_lds(g0, g1, z4, z4, 0);
#endif
}
#endif

// ---------------- prep kernels (once) ----------------

__global__ void k_zero(float* __restrict__ h, float* __restrict__ msg,
                       _Float16* __restrict__ h16) {
    int idx = blockIdx.x * blockDim.x + threadIdx.x;
    if (idx < NN * SD) { h[idx] = 0.0f; h16[idx] = (_Float16)0.0f; }
    if (idx < NN * MD) msg[idx] = 0.0f;
}

__global__ void k_prep_bias(const float* __restrict__ b, const float* __restrict__ Wmsg,
                            const float* __restrict__ bmsg,
                            float* __restrict__ aib, float* __restrict__ ajb) {
    int idx = blockIdx.x * blockDim.x + threadIdx.x;
    if (idx >= NN * MD) return;
    int i = idx >> 4, k = idx & 15;
    float wbi = Wmsg[k * 67 + 2 * SD + 1];
    float wbj = Wmsg[k * 67 + 2 * SD + 2];
    aib[idx] = b[i] * wbi + bmsg[k];
    ajb[idx] = b[i] * wbj;
}

__global__ void k_prep_weights(const float* __restrict__ Wmsg, const float* __restrict__ Wih,
                               _Float16* __restrict__ Whi, _Float16* __restrict__ Whj,
                               _Float16* __restrict__ WihB) {
    for (int idx = threadIdx.x; idx < MD * SD; idx += blockDim.x) {
        int n = idx / SD, k = idx % SD;
        Whi[idx] = (_Float16)Wmsg[n * 67 + k];
        Whj[idx] = (_Float16)Wmsg[n * 67 + SD + k];
    }
    // W_ih as B-matrix: [XKP x GD], B[k][n] = W_ih[n][k], zero-padded k>=48
    for (int idx = threadIdx.x; idx < XKP * GD; idx += blockDim.x) {
        int k = idx / GD, n = idx % GD;
        WihB[idx] = (k < SD + MD) ? (_Float16)Wih[n * (SD + MD) + k] : (_Float16)0.0f;
    }
}

// ---------------- WMMA: ai = h @ Whi^T + aib ; aj = h @ Whj^T + ajb ----------------

__global__ __launch_bounds__(32)
void k_aiaj(const _Float16* __restrict__ h16,
            const _Float16* __restrict__ Whi, const _Float16* __restrict__ Whj,
            const float* __restrict__ aib, const float* __restrict__ ajb,
            float* __restrict__ ai, float* __restrict__ aj) {
    const int lane  = threadIdx.x;
    const int m0    = blockIdx.x * 16;
    const int row   = m0 + (lane & 15);
    const int abase = (lane & 16) ? 16 : 0;  // A: lanes 16-31 hold K=16..31
    V16H a;
#pragma unroll
    for (int e = 0; e < 16; ++e) {
        int k = abase + e;
        a.e[e] = h16[row * SD + k];
    }
    const int n  = lane & 15;                // B/C/D: lane -> N column
    const int kb = (lane & 16) ? 16 : 0;     // B: lanes 16-31 hold K=16..31
    V16H bi, bj;
#pragma unroll
    for (int e = 0; e < 16; ++e) {
        bi.e[e] = Whi[n * SD + kb + e];
        bj.e[e] = Whj[n * SD + kb + e];
    }
    V8F ci = {}, cj = {};
    ci.v = __builtin_amdgcn_wmma_f32_16x16x32_f16(false, a.v, false, bi.v, (short)0, ci.v, false, false);
    cj.v = __builtin_amdgcn_wmma_f32_16x16x32_f16(false, a.v, false, bj.v, (short)0, cj.v, false, false);
#pragma unroll
    for (int r = 0; r < 8; ++r) {
        int orow = m0 + r + ((lane & 16) ? 8 : 0);
        ai[orow * MD + n] = ci.e[r] + aib[orow * MD + n];
        aj[orow * MD + n] = cj.e[r] + ajb[orow * MD + n];
    }
}

// ---- message pass: msg[j,k] += sum_i relu(ai[i,k]+aj[j,k]+J[i,j]*wJ[k]) ----
// block = 256 threads = 16 j-cols x 16 k.  J (16MB) is L2-resident (192MB L2),
// so the loop is VALU-bound; data movement goes through the Tensor Data Mover:
// wave 0 DMAs the next 128x16 J tile + 128x16 ai chunk into LDS (double-
// buffered) while all 8 waves burn VALU on the current buffer.

__global__ __launch_bounds__(256)
void k_msg(const float* __restrict__ J, const float* __restrict__ Wmsg,
           const float* __restrict__ ai, const float* __restrict__ aj,
           float* __restrict__ msg) {
    __shared__ float Js[2][CH * 16];
    __shared__ float As[2][CH * 16];
    const int tid = threadIdx.x;
    const int k   = tid & 15;
    const int jj  = tid >> 4;
    const int j0  = blockIdx.x * 16;
    const int j   = j0 + jj;
    const float ajv = aj[j * MD + k];
    const float wjk = Wmsg[k * 67 + 2 * SD];
    float acc = 0.0f;

#ifdef HAVE_TDM
    if (tid < 32) {   // wave 0 owns the DMA engine for this block
        tdm_load_2d((unsigned int)(size_t)&Js[0][0], &J[(size_t)0 * NN + j0],
                    16u, (unsigned int)CH, (unsigned long long)NN, NN, NN);
        tdm_load_2d((unsigned int)(size_t)&As[0][0], &ai[0],
                    (unsigned int)(CH * MD), 1u, (unsigned long long)(CH * MD), NN * MD, 1);
        __builtin_amdgcn_s_wait_tensorcnt(0);
    }
    __syncthreads();
    int buf = 0;
    for (int i0 = 0; i0 < NN; i0 += CH) {
        const int inext = i0 + CH;
        if (inext < NN && tid < 32) {
            tdm_load_2d((unsigned int)(size_t)&Js[buf ^ 1][0], &J[(size_t)inext * NN + j0],
                        16u, (unsigned int)CH, (unsigned long long)NN, NN, NN);
            tdm_load_2d((unsigned int)(size_t)&As[buf ^ 1][0], &ai[inext * MD],
                        (unsigned int)(CH * MD), 1u, (unsigned long long)(CH * MD), NN * MD, 1);
        }
        const float* __restrict__ Jb = Js[buf];
        const float* __restrict__ Ab = As[buf];
#pragma unroll 8
        for (int ii = 0; ii < CH; ++ii) {
            float v = fmaf(Jb[ii * 16 + jj], wjk, Ab[ii * 16 + k] + ajv);
            acc += fmaxf(v, 0.0f);
        }
        if (tid < 32) __builtin_amdgcn_s_wait_tensorcnt(0);
        __syncthreads();
        buf ^= 1;
    }
#else
    for (int i0 = 0; i0 < NN; i0 += CH) {
        __syncthreads();
#pragma unroll
        for (int l = tid; l < CH * 16; l += 256) {
            Js[0][l] = J[(size_t)(i0 + (l >> 4)) * NN + j0 + (l & 15)];
            As[0][l] = ai[i0 * MD + l];
        }
        __syncthreads();
#pragma unroll 8
        for (int ii = 0; ii < CH; ++ii) {
            float v = fmaf(Js[0][ii * 16 + jj], wjk, As[0][ii * 16 + k] + ajv);
            acc += fmaxf(v, 0.0f);
        }
    }
#endif
    msg[j * MD + k] += acc;
}

// ---------------- WMMA: g = [h|msg] @ W_ih^T + b_ih  (K=48 padded to 64) ----------------

__global__ __launch_bounds__(32)
void k_gemm_g(const float* __restrict__ h, const float* __restrict__ msg,
              const _Float16* __restrict__ WihB, const float* __restrict__ bih,
              float* __restrict__ g) {
    const int lane  = threadIdx.x;
    const int m0    = blockIdx.x * 16;
    const int n0    = blockIdx.y * 16;
    const int row   = m0 + (lane & 15);
    const int abase = (lane & 16) ? 16 : 0;
    const int ncol  = n0 + (lane & 15);
    const int kb    = (lane & 16) ? 16 : 0;
    V8F c = {};
#pragma unroll
    for (int s = 0; s < 2; ++s) {
        const int kbase = s * 32;
        V16H a, bv;
#pragma unroll
        for (int e = 0; e < 16; ++e) {
            int kk = kbase + abase + e;
            float xv;
            if (kk < SD)           xv = h[row * SD + kk];
            else if (kk < SD + MD) xv = msg[row * MD + (kk - SD)];
            else                   xv = 0.0f;
            a.e[e]  = (_Float16)xv;
            bv.e[e] = WihB[(kbase + kb + e) * GD + ncol];
        }
        c.v = __builtin_amdgcn_wmma_f32_16x16x32_f16(false, a.v, false, bv.v, (short)0, c.v, false, false);
    }
#pragma unroll
    for (int r = 0; r < 8; ++r) {
        int orow = m0 + r + ((lane & 16) ? 8 : 0);
        g[orow * GD + ncol] = c.e[r] + bih[ncol];
    }
}

// ---------------- gates: h = (1-z)*n  (also emits f16 copy for next step's WMMA) -------

__global__ void k_gates(const float* __restrict__ g, const float* __restrict__ bhh,
                        float* __restrict__ h, _Float16* __restrict__ h16) {
    int idx = blockIdx.x * blockDim.x + threadIdx.x;
    if (idx >= NN * SD) return;
    int row = idx >> 5, u = idx & 31;
    float gr = g[row * GD + u];
    float gz = g[row * GD + SD + u];
    float gn = g[row * GD + 2 * SD + u];
    float r = sigmoid_f(gr + bhh[u]);
    float z = sigmoid_f(gz + bhh[SD + u]);
    float n = tanhf(gn + r * bhh[2 * SD + u]);
    float hv = (1.0f - z) * n;
    h[idx]   = hv;
    h16[idx] = (_Float16)hv;
}

// ---------------- readout + softmax over N=2048 (single block) ----------------

__global__ __launch_bounds__(1024)
void k_softmax(const float* __restrict__ h, const float* __restrict__ Wro,
               const float* __restrict__ bro, float* __restrict__ out) {
    __shared__ float red[1024];
    const int t = threadIdx.x;
    float l0 = bro[0], l1 = bro[0];
#pragma unroll
    for (int u = 0; u < SD; ++u) {
        float w = Wro[u];
        l0 += h[t * SD + u] * w;
        l1 += h[(t + 1024) * SD + u] * w;
    }
    red[t] = fmaxf(l0, l1);
    __syncthreads();
    for (int s = 512; s > 0; s >>= 1) {
        if (t < s) red[t] = fmaxf(red[t], red[t + s]);
        __syncthreads();
    }
    float m = red[0];
    __syncthreads();
    float e0 = __expf(l0 - m), e1 = __expf(l1 - m);
    red[t] = e0 + e1;
    __syncthreads();
    for (int s = 512; s > 0; s >>= 1) {
        if (t < s) red[t] += red[t + s];
        __syncthreads();
    }
    float inv = 1.0f / red[0];
    out[t] = e0 * inv;
    out[t + 1024] = e1 * inv;
}

// ---------------- host launch ----------------

extern "C" void kernel_launch(void* const* d_in, const int* in_sizes, int n_in,
                              void* d_out, int out_size, void* d_ws, size_t ws_size,
                              hipStream_t stream) {
    (void)in_sizes; (void)n_in; (void)out_size; (void)ws_size;
    const float* J    = (const float*)d_in[0];
    const float* b    = (const float*)d_in[1];
    const float* Wmsg = (const float*)d_in[2];
    const float* bmsg = (const float*)d_in[3];
    const float* Wih  = (const float*)d_in[4];
    // d_in[5] = W_hh: unused by the reference math (only b_hh enters the gates)
    const float* bih  = (const float*)d_in[6];
    const float* bhh  = (const float*)d_in[7];
    const float* Wro  = (const float*)d_in[8];
    const float* bro  = (const float*)d_in[9];
    float* out = (float*)d_out;

    char* p = (char*)d_ws;
    auto carve = [&](size_t bytes) {
        char* q = p;
        p += (bytes + 255) & ~(size_t)255;
        return q;
    };
    float*    h    = (float*)carve((size_t)NN * SD * 4);
    float*    msg  = (float*)carve((size_t)NN * MD * 4);
    float*    ai   = (float*)carve((size_t)NN * MD * 4);
    float*    aj   = (float*)carve((size_t)NN * MD * 4);
    float*    aib  = (float*)carve((size_t)NN * MD * 4);
    float*    ajb  = (float*)carve((size_t)NN * MD * 4);
    float*    g    = (float*)carve((size_t)NN * GD * 4);
    _Float16* h16  = (_Float16*)carve((size_t)NN * SD * 2);
    _Float16* Whi  = (_Float16*)carve((size_t)MD * SD * 2);
    _Float16* Whj  = (_Float16*)carve((size_t)MD * SD * 2);
    _Float16* WihB = (_Float16*)carve((size_t)XKP * GD * 2);

    k_zero<<<(NN * SD + 255) / 256, 256, 0, stream>>>(h, msg, h16);
    k_prep_bias<<<(NN * MD + 255) / 256, 256, 0, stream>>>(b, Wmsg, bmsg, aib, ajb);
    k_prep_weights<<<1, 256, 0, stream>>>(Wmsg, Wih, Whi, Whj, WihB);

    for (int s = 0; s < STEPS; ++s) {
        k_aiaj<<<NN / 16, 32, 0, stream>>>(h16, Whi, Whj, aib, ajb, ai, aj);
        k_msg<<<NN / 16, 256, 0, stream>>>(J, Wmsg, ai, aj, msg);
        dim3 gg(NN / 16, GD / 16);
        k_gemm_g<<<gg, 32, 0, stream>>>(h, msg, WihB, bih, g);
        k_gates<<<(NN * SD + 255) / 256, 256, 0, stream>>>(g, bhh, h, h16);
    }
    k_softmax<<<1, 1024, 0, stream>>>(h, Wro, bro, out);
}
